// CTI_toC_36438502539817
// MI455X (gfx1250) — matrix operand order repaired and checked by reading, not compiled, not run on hardware
//
#include <hip/hip_runtime.h>
#include <hip/hip_bf16.h>

// ---------------------------------------------------------------------------
// CDNA5 (gfx1250) implementation of the CTI block:
//   prep+LN -> WMMA GEMMs (value/off/att) -> MSDA gather -> WMMA GEMM (Wout,
//   +residual) -> LN -> WMMA GEMM (fc1) -> dwconv3x3+GELU -> WMMA GEMM (fc2,
//   +residual into d_out).
// GEMMs: v_wmma_f32_16x16x32_bf16, wave32; each wave owns a 16x32 tile (A-frag
// reused across two WMMAs); the 32-col weight tile is staged in LDS via
// global_load_async_to_lds_b128 (s_wait_asynccnt); B fragments read back as
// ds_load_b128 (uint4).
// ---------------------------------------------------------------------------

typedef __attribute__((ext_vector_type(16))) __bf16 v16bf;
typedef __attribute__((ext_vector_type(8)))  float  v8f;
typedef int v4i_vs __attribute__((vector_size(4 * sizeof(int))));

#define DIMC   384
#define HEADS  6
#define HDIM   64
#define NTOK   12096      // tokens per batch (21*n, n=576)
#define MROWS  24192      // B*NTOK
#define L0N    9216       // 96*96
#define L1N    2304       // 48*48
#define L1END  11520
#define OFF_LD 160        // Woff N padded 144 -> 160 (for 32-wide tiles)
#define ATT_LD 96         // Watt N padded 72  -> 96

#if __has_builtin(__builtin_amdgcn_global_load_async_to_lds_b128) && \
    __has_builtin(__builtin_amdgcn_s_wait_asynccnt)
#define USE_ASYNC_LDS 1
#else
#define USE_ASYNC_LDS 0
#endif

__device__ __forceinline__ unsigned short f2bf(float f) {
    unsigned int u = __float_as_uint(f);
    unsigned int r = (u + 0x7FFFu + ((u >> 16) & 1u)) >> 16;
    return (unsigned short)r;
}

__device__ __forceinline__ void tok_pos(int t, int& y, int& x, int& Hl, int& Wl, int& st) {
    if (t < L0N)        { Hl = 96; Wl = 96; st = 0;     y = t / 96;        x = t - y * 96; }
    else if (t < L1END) { int i = t - L0N;   Hl = 48; Wl = 48; st = L0N;   y = i / 48; x = i - y * 48; }
    else                { int i = t - L1END; Hl = 24; Wl = 24; st = L1END; y = i / 24; x = i - y * 24; }
}

// ---------------------------------------------------------------------------
// Weight convert: W (K x N, f32, row-major) -> Wt (Npad x K, bf16) == B^T
// padded columns are zero so padded WMMA tiles contribute nothing.
// ---------------------------------------------------------------------------
__global__ void wt_convert_kernel(const float* __restrict__ W, unsigned short* __restrict__ Wt,
                                  int K, int N, int Npad) {
    int i = blockIdx.x * blockDim.x + threadIdx.x;
    if (i >= Npad * K) return;
    int n = i / K, k = i - n * K;
    Wt[i] = (n < N) ? f2bf(W[(size_t)k * N + n]) : (unsigned short)0;
}

// ---------------------------------------------------------------------------
// block (128 thr) reduction of two values
// ---------------------------------------------------------------------------
__device__ __forceinline__ void block_reduce2(float& a, float& b, float* s1, float* s2, int tid) {
    s1[tid] = a; s2[tid] = b; __syncthreads();
#pragma unroll
    for (int o = 64; o > 0; o >>= 1) {
        if (tid < o) { s1[tid] += s1[tid + o]; s2[tid] += s2[tid + o]; }
        __syncthreads();
    }
    a = s1[0]; b = s2[0];
    __syncthreads();
}

// ---------------------------------------------------------------------------
// prep: q = query (+feat on level-1 tokens); qn = LN(q, cti_q); fn = LN(q, cti_f)
//       aq = LN(qn, cf_q) -> bf16 ; af = LN(fn, cf_f) -> bf16 ; qn kept f32.
// one block per token row, 128 threads x 3 elements.
// ---------------------------------------------------------------------------
__global__ void prep_ln_kernel(const float* __restrict__ query, const float* __restrict__ feat,
                               const float* __restrict__ qw, const float* __restrict__ qb,
                               const float* __restrict__ fw, const float* __restrict__ fb,
                               const float* __restrict__ cqw, const float* __restrict__ cqb,
                               const float* __restrict__ cfw, const float* __restrict__ cfb,
                               float* __restrict__ qn_out,
                               unsigned short* __restrict__ aq_out,
                               unsigned short* __restrict__ af_out) {
    __shared__ float s1[128], s2[128];
    int r = blockIdx.x, tid = threadIdx.x;
    int b = r / NTOK, t = r - b * NTOK;
    bool addf = (t >= L0N) && (t < L1END);
    const float* qrow = query + (size_t)r * DIMC;
    const float* frow = feat + (size_t)(b * L1N + (t - L0N)) * DIMC;

    float x[3]; float s = 0.f, ss = 0.f;
#pragma unroll
    for (int i = 0; i < 3; i++) {
        int idx = tid + i * 128;
        float v = qrow[idx];
        if (addf) v += frow[idx];
        x[i] = v; s += v; ss += v * v;
    }
    block_reduce2(s, ss, s1, s2, tid);
    float m = s * (1.f / DIMC);
    float inv = rsqrtf(ss * (1.f / DIMC) - m * m + 1e-6f);

    float qn[3], fn[3];
    float sq = 0.f, ssq = 0.f, sf = 0.f, ssf = 0.f;
#pragma unroll
    for (int i = 0; i < 3; i++) {
        int idx = tid + i * 128;
        float xc = (x[i] - m) * inv;
        qn[i] = xc * qw[idx] + qb[idx];
        fn[i] = xc * fw[idx] + fb[idx];
        sq += qn[i]; ssq += qn[i] * qn[i];
        sf += fn[i]; ssf += fn[i] * fn[i];
    }
    block_reduce2(sq, ssq, s1, s2, tid);
    float mq = sq * (1.f / DIMC);
    float invq = rsqrtf(ssq * (1.f / DIMC) - mq * mq + 1e-6f);
    block_reduce2(sf, ssf, s1, s2, tid);
    float mf = sf * (1.f / DIMC);
    float invf = rsqrtf(ssf * (1.f / DIMC) - mf * mf + 1e-6f);

#pragma unroll
    for (int i = 0; i < 3; i++) {
        int idx = tid + i * 128;
        size_t o = (size_t)r * DIMC + idx;
        qn_out[o] = qn[i];
        aq_out[o] = f2bf((qn[i] - mq) * invq * cqw[idx] + cqb[idx]);
        af_out[o] = f2bf((fn[i] - mf) * invf * cfw[idx] + cfb[idx]);
    }
}

// ---------------------------------------------------------------------------
// LN (single) -> bf16, for FFN input.
// ---------------------------------------------------------------------------
__global__ void ln_bf16_kernel(const float* __restrict__ X,
                               const float* __restrict__ w, const float* __restrict__ bb,
                               unsigned short* __restrict__ out) {
    __shared__ float s1[128], s2[128];
    int r = blockIdx.x, tid = threadIdx.x;
    const float* row = X + (size_t)r * DIMC;
    float x[3]; float s = 0.f, ss = 0.f;
#pragma unroll
    for (int i = 0; i < 3; i++) {
        int idx = tid + i * 128;
        float v = row[idx];
        x[i] = v; s += v; ss += v * v;
    }
    block_reduce2(s, ss, s1, s2, tid);
    float m = s * (1.f / DIMC);
    float inv = rsqrtf(ss * (1.f / DIMC) - m * m + 1e-6f);
#pragma unroll
    for (int i = 0; i < 3; i++) {
        int idx = tid + i * 128;
        out[(size_t)r * DIMC + idx] = f2bf((x[i] - m) * inv * w[idx] + bb[idx]);
    }
}

// ---------------------------------------------------------------------------
// WMMA bf16 GEMM: C[M x ldc] (+bias, +res) = A[M x K] * Bt^T  (Bt is Npad x K)
// 256 threads = 8 waves; each wave computes a 16x32 tile (two accumulators,
// A-fragment reused). Block covers 128 rows x 32 cols; grid (Npad/32, M/128).
// The 32 x K bf16 weight tile is staged in LDS (async-to-LDS when available);
// B fragments are read back as 16B ds_load_b128.
// Fragment layouts follow CDNA5 ISA 7.12.2 (16-bit A 16x32, B 32x16, f32 C/D).
// ---------------------------------------------------------------------------
__global__ void gemm_wmma_kernel(const unsigned short* __restrict__ A,
                                 const unsigned short* __restrict__ Bt,
                                 const float* __restrict__ bias,
                                 const float* __restrict__ res,
                                 float* __restrict__ C,
                                 int K, int ldc, int Nstore) {
    __shared__ unsigned short btile[32 * DIMC];   // up to 24 KB (of 320 KB/WGP)

    const int tid  = threadIdx.x;
    const int lane = tid & 31;
    const int wave = tid >> 5;
    const int n0b  = blockIdx.x * 32;
    const int m0   = blockIdx.y * 128 + wave * 16;
    const int l16  = lane & 15;
    const bool hi  = lane >= 16;

    // ---- stage B tile: cols [n0b, n0b+32) x K, contiguous 32*K*2 bytes ----
    {
        const int chunks = 4 * K;                 // 16-byte chunks
        const char* gsrc = (const char*)(Bt + (size_t)n0b * K);
        char* ldst = (char*)btile;
#if USE_ASYNC_LDS
        for (int c = tid; c < chunks; c += 256) {
            __builtin_amdgcn_global_load_async_to_lds_b128(
                (__attribute__((address_space(1))) v4i_vs*)(gsrc + (size_t)c * 16),
                (__attribute__((address_space(3))) v4i_vs*)(ldst + (size_t)c * 16),
                0, 0);
        }
        __builtin_amdgcn_s_wait_asynccnt(0);
#else
        const uint4* gs = (const uint4*)gsrc;
        uint4* ld = (uint4*)ldst;
        for (int c = tid; c < chunks; c += 256) ld[c] = gs[c];
#endif
    }
    __syncthreads();

    const unsigned int* Arow = (const unsigned int*)(A + (size_t)(m0 + l16) * K);
    const uint4* B0q = (const uint4*)(btile + (size_t)l16 * K);        // 16B units
    const uint4* B1q = (const uint4*)(btile + (size_t)(16 + l16) * K);
    const int ka = hi ? 4 : 0;   // +8 bf16 = +4 dwords (A upper-lane K offset)
    const int kq = hi ? 1 : 0;   // +8 bf16 = +1 uint4 (B upper-lane K offset)

    v8f acc0 = {}, acc1 = {};
    for (int k0 = 0; k0 < K; k0 += 32) {
        const int base = k0 >> 1;       // dword units
        const int baseq = k0 >> 3;      // uint4 units
        union { unsigned int u[8]; uint4 q[2]; v16bf v; } a, b0, b1;
#pragma unroll
        for (int j = 0; j < 4; j++) {
            a.u[j]     = Arow[base + ka + j];        // K = k0 + {0,2,4,6}(+8)
            a.u[4 + j] = Arow[base + ka + 8 + j];    // K = k0 + 16 + {0,2,4,6}(+8)
        }
        b0.q[0] = B0q[baseq + kq];                   // K = k0 + 0..7, 16..23  (+8)
        b0.q[1] = B0q[baseq + kq + 2];
        b1.q[0] = B1q[baseq + kq];
        b1.q[1] = B1q[baseq + kq + 2];
        acc0 = __builtin_amdgcn_wmma_f32_16x16x32_bf16(
            false, a.v, false, b0.v, (short)0, acc0, false, false);
        acc1 = __builtin_amdgcn_wmma_f32_16x16x32_bf16(
            false, a.v, false, b1.v, (short)0, acc1, false, false);
    }

    const int rbase = m0 + (hi ? 8 : 0);
    const int nc0 = n0b + l16;
    const int nc1 = n0b + 16 + l16;
    if (nc0 < Nstore) {
        const float bv = bias ? bias[nc0] : 0.f;
#pragma unroll
        for (int v = 0; v < 8; v++) {
            size_t o = (size_t)(rbase + v) * ldc + nc0;
            float val = acc0[v] + bv;
            if (res) val += res[o];
            C[o] = val;
        }
    }
    if (nc1 < Nstore) {
        const float bv = bias ? bias[nc1] : 0.f;
#pragma unroll
        for (int v = 0; v < 8; v++) {
            size_t o = (size_t)(rbase + v) * ldc + nc1;
            float val = acc1[v] + bv;
            if (res) val += res[o];
            C[o] = val;
        }
    }
}

// ---------------------------------------------------------------------------
// MSDA: one wave per (row, head); lane handles 2 of 64 head channels (float2).
// softmax over 12 logits, 3 levels x 4 points x 4 bilinear taps; bf16 output.
// ---------------------------------------------------------------------------
__global__ void msda_kernel(const float* __restrict__ value, const float* __restrict__ off,
                            const float* __restrict__ att, unsigned short* __restrict__ attn) {
    const int lane = threadIdx.x & 31;
    const int wid = (blockIdx.x << 3) + (threadIdx.x >> 5);
    if (wid >= MROWS * HEADS) return;
    const int h = wid % HEADS;
    const int r = wid / HEADS;
    const int b = r / NTOK, t = r - b * NTOK;

    int ty, tx, tH, tW, tst;
    tok_pos(t, ty, tx, tH, tW, tst);
    const float refx = (tx + 0.5f) / tW;
    const float refy = (ty + 0.5f) / tH;

    const float* lg = att + (size_t)r * ATT_LD + h * 12;
    float w[12];
    float mx = lg[0];
#pragma unroll
    for (int j = 1; j < 12; j++) mx = fmaxf(mx, lg[j]);
    float sum = 0.f;
#pragma unroll
    for (int j = 0; j < 12; j++) { w[j] = __expf(lg[j] - mx); sum += w[j]; }
    const float invs = 1.f / sum;

    const float* offp = off + (size_t)r * OFF_LD + h * 24;
    const int ch = lane * 2;
    float a0 = 0.f, a1 = 0.f;
    const int Hls[3] = {96, 48, 24};
    const int sts[3] = {0, L0N, L1END};

#pragma unroll
    for (int l = 0; l < 3; l++) {
        const int Hl = Hls[l], Wl = Hls[l], st = sts[l];
        const float* vb = value + (size_t)(b * NTOK + st) * DIMC + h * HDIM + ch;
#pragma unroll
        for (int p = 0; p < 4; p++) {
            const float ox = offp[l * 8 + p * 2];
            const float oy = offp[l * 8 + p * 2 + 1];
            const float xx = (refx + ox / Wl) * Wl - 0.5f;
            const float yy = (refy + oy / Hl) * Hl - 0.5f;
            const float x0f = floorf(xx), y0f = floorf(yy);
            const int x0 = (int)x0f, y0 = (int)y0f;
            const float fx = xx - x0f, fy = yy - y0f;
            float s0 = 0.f, s1 = 0.f;
            auto tap = [&](int yi, int xi, float wt) {
                if (yi >= 0 && yi < Hl && xi >= 0 && xi < Wl) {
                    const float2 v2 = *(const float2*)(vb + (size_t)(yi * Wl + xi) * DIMC);
                    s0 += v2.x * wt; s1 += v2.y * wt;
                }
            };
            tap(y0,     x0,     (1.f - fx) * (1.f - fy));
            tap(y0,     x0 + 1, fx * (1.f - fy));
            tap(y0 + 1, x0,     (1.f - fx) * fy);
            tap(y0 + 1, x0 + 1, fx * fy);
            const float aw = w[l * 4 + p] * invs;
            a0 += aw * s0; a1 += aw * s1;
        }
    }
    const size_t o = (size_t)r * DIMC + h * HDIM + ch;
    attn[o]     = f2bf(a0);
    attn[o + 1] = f2bf(a1);
}

// ---------------------------------------------------------------------------
// per-level depthwise 3x3 (SAME, zero pad) + exact GELU -> bf16
// ---------------------------------------------------------------------------
__global__ void dwconv_gelu_kernel(const float* __restrict__ hbuf,
                                   const float* __restrict__ dww, const float* __restrict__ dwb,
                                   unsigned short* __restrict__ out) {
    int i = blockIdx.x * blockDim.x + threadIdx.x;
    if (i >= MROWS * 96) return;
    const int ch = i % 96;
    const int r = i / 96;
    const int b = r / NTOK, t = r - b * NTOK;
    int y, x, Hl, Wl, st;
    tok_pos(t, y, x, Hl, Wl, st);
    float acc = dwb[ch];
#pragma unroll
    for (int ky = -1; ky <= 1; ky++) {
#pragma unroll
        for (int kx = -1; kx <= 1; kx++) {
            const int yy = y + ky, xxp = x + kx;
            if (yy >= 0 && yy < Hl && xxp >= 0 && xxp < Wl) {
                acc += hbuf[(size_t)(b * NTOK + st + yy * Wl + xxp) * 96 + ch] *
                       dww[((ky + 1) * 3 + (kx + 1)) * 96 + ch];
            }
        }
    }
    const float g = 0.5f * acc * (1.f + erff(acc * 0.70710678118654752f));
    out[i] = f2bf(g);
}

// ---------------------------------------------------------------------------
extern "C" void kernel_launch(void* const* d_in, const int* in_sizes, int n_in,
                              void* d_out, int out_size, void* d_ws, size_t ws_size,
                              hipStream_t stream) {
    (void)in_sizes; (void)n_in; (void)out_size; (void)ws_size;
    const float* query  = (const float*)d_in[0];
    const float* feat   = (const float*)d_in[2];
    const float* ctiqw  = (const float*)d_in[7];
    const float* ctiqb  = (const float*)d_in[8];
    const float* ctifw  = (const float*)d_in[9];
    const float* ctifb  = (const float*)d_in[10];
    const float* cfqw   = (const float*)d_in[11];
    const float* cfqb   = (const float*)d_in[12];
    const float* cffw   = (const float*)d_in[13];
    const float* cffb   = (const float*)d_in[14];
    const float* Wv     = (const float*)d_in[15];
    const float* bv     = (const float*)d_in[16];
    const float* Woff   = (const float*)d_in[17];
    const float* boff   = (const float*)d_in[18];
    const float* Watt   = (const float*)d_in[19];
    const float* batt   = (const float*)d_in[20];
    const float* Wout   = (const float*)d_in[21];
    const float* bout   = (const float*)d_in[22];
    const float* ffnw   = (const float*)d_in[23];
    const float* ffnb   = (const float*)d_in[24];
    const float* fc1w   = (const float*)d_in[25];
    const float* fc1b   = (const float*)d_in[26];
    const float* dww    = (const float*)d_in[27];
    const float* dwb    = (const float*)d_in[28];
    const float* fc2w   = (const float*)d_in[29];
    const float* fc2b   = (const float*)d_in[30];
    float* outp = (float*)d_out;

    // ---- workspace layout (buffers reused across phases) ----
    char* ws = (char*)d_ws;
    size_t o = 0;
    float* qn_f32 = (float*)(ws + o);          o += (size_t)MROWS * DIMC * 4;
    unsigned short* aq_bf = (unsigned short*)(ws + o); o += (size_t)MROWS * DIMC * 2; // -> ln_ffn
    unsigned short* af_bf = (unsigned short*)(ws + o); o += (size_t)MROWS * DIMC * 2; // -> attn
    float* val_f32 = (float*)(ws + o);         o += (size_t)MROWS * DIMC * 4;
    float* off_f32 = (float*)(ws + o);         o += (size_t)MROWS * OFF_LD * 4;       // -> h f32 (M x 96)
    float* att_f32 = (float*)(ws + o);         o += (size_t)MROWS * ATT_LD * 4;       // -> gelu bf16 (M x 96)
    unsigned short* wv_t   = (unsigned short*)(ws + o); o += (size_t)384 * 384 * 2;
    unsigned short* woff_t = (unsigned short*)(ws + o); o += (size_t)OFF_LD * 384 * 2;
    unsigned short* watt_t = (unsigned short*)(ws + o); o += (size_t)ATT_LD * 384 * 2;
    unsigned short* wout_t = (unsigned short*)(ws + o); o += (size_t)384 * 384 * 2;
    unsigned short* fc1_t  = (unsigned short*)(ws + o); o += (size_t)96 * 384 * 2;
    unsigned short* fc2_t  = (unsigned short*)(ws + o); o += (size_t)384 * 96 * 2;

    unsigned short* attn_bf  = af_bf;     // reuse after value GEMM consumed af
    unsigned short* lnffn_bf = aq_bf;     // reuse after off/att GEMMs consumed aq
    float* h_f32 = off_f32;               // reuse after MSDA consumed offsets
    unsigned short* hg_bf = (unsigned short*)att_f32; // reuse after MSDA consumed logits

    // ---- 1) weights -> bf16 transposed (zero-padded to 32-col multiples) ----
    auto cvt = [&](const float* W, unsigned short* Wt, int K, int N, int Npad) {
        int total = Npad * K;
        wt_convert_kernel<<<(total + 255) / 256, 256, 0, stream>>>(W, Wt, K, N, Npad);
    };
    cvt(Wv,   wv_t,   384, 384, 384);
    cvt(Woff, woff_t, 384, 144, OFF_LD);
    cvt(Watt, watt_t, 384, 72,  ATT_LD);
    cvt(Wout, wout_t, 384, 384, 384);
    cvt(fc1w, fc1_t,  384, 96,  96);
    cvt(fc2w, fc2_t,  96,  384, 384);

    // ---- 2) q build + stacked LayerNorms ----
    prep_ln_kernel<<<MROWS, 128, 0, stream>>>(query, feat, ctiqw, ctiqb, ctifw, ctifb,
                                              cfqw, cfqb, cffw, cffb,
                                              qn_f32, aq_bf, af_bf);

    const dim3 blk(256);
    const int mBlocks = MROWS / 128;  // 189

    // ---- 3) projections (WMMA) ----
    gemm_wmma_kernel<<<dim3(384 / 32, mBlocks), blk, 0, stream>>>(
        af_bf, wv_t, bv, nullptr, val_f32, 384, 384, 384);
    gemm_wmma_kernel<<<dim3(OFF_LD / 32, mBlocks), blk, 0, stream>>>(
        aq_bf, woff_t, boff, nullptr, off_f32, 384, OFF_LD, 144);
    gemm_wmma_kernel<<<dim3(ATT_LD / 32, mBlocks), blk, 0, stream>>>(
        aq_bf, watt_t, batt, nullptr, att_f32, 384, ATT_LD, 72);

    // ---- 4) deformable sampling ----
    msda_kernel<<<(MROWS * HEADS) / 8, blk, 0, stream>>>(val_f32, off_f32, att_f32, attn_bf);

    // ---- 5) output projection + residual (writes d_out) ----
    gemm_wmma_kernel<<<dim3(384 / 32, mBlocks), blk, 0, stream>>>(
        attn_bf, wout_t, bout, qn_f32, outp, 384, 384, 384);

    // ---- 6) FFN ----
    ln_bf16_kernel<<<MROWS, 128, 0, stream>>>(outp, ffnw, ffnb, lnffn_bf);
    gemm_wmma_kernel<<<dim3(96 / 32, mBlocks), blk, 0, stream>>>(
        lnffn_bf, fc1_t, fc1b, nullptr, h_f32, 384, 96, 96);
    dwconv_gelu_kernel<<<(MROWS * 96 + 255) / 256, blk, 0, stream>>>(h_f32, dww, dwb, hg_bf);
    gemm_wmma_kernel<<<dim3(384 / 32, mBlocks), blk, 0, stream>>>(
        hg_bf, fc2_t, fc2b, outp, outp, 96, 384, 384);
}
